// T5Attention_36051955483033
// MI455X (gfx1250) — matrix operand order repaired and checked by reading, not compiled, not run on hardware
//
#include <hip/hip_runtime.h>
#include <hip/hip_bf16.h>

#define D_MODEL 1024
#define N_HEADS 16
#define D_KV    64
#define INNER   1024
#define BATCH   4
#define SEQ     2048
#define ROWS    (BATCH * SEQ)   // 8192

typedef __bf16 bf16_t;
typedef __attribute__((ext_vector_type(8)))  __bf16 v8bf;
typedef __attribute__((ext_vector_type(16))) __bf16 v16bf;
typedef __attribute__((ext_vector_type(8)))  float  v8f;

// ---------------- TDM (Tensor Data Mover) staging ----------------
#if defined(__HIP_DEVICE_COMPILE__) && __has_builtin(__builtin_amdgcn_tensor_load_to_lds)
#define HAVE_TDM 1
#else
#define HAVE_TDM 0
#endif

#if HAVE_TDM
typedef __attribute__((ext_vector_type(4))) unsigned int tdm_g0_t;
typedef __attribute__((ext_vector_type(8))) int          tdm_g1_t;
typedef __attribute__((ext_vector_type(4))) int          tdm_g2_t;

// DMA a [th][tw] bf16 tile (row stride `stride_elems` in global) into LDS,
// packed contiguously row-major at lds_ptr. D# layout per CDNA5 ISA section 8.
__device__ __forceinline__ void tdm_load_2d(void* lds_ptr, const void* gptr,
                                            unsigned tw, unsigned th_,
                                            unsigned stride_elems) {
  unsigned long long ga = (unsigned long long)gptr;
  unsigned lds = (unsigned)(unsigned long long)lds_ptr;   // LDS byte address = low 32 bits
  tdm_g0_t g0;
  g0[0] = 1u;                                  // count=1 (valid user descriptor)
  g0[1] = lds;                                 // lds_addr
  g0[2] = (unsigned)ga;                        // global_addr[31:0]
  g0[3] = (unsigned)(ga >> 32) | (2u << 30);   // global_addr[56:32] | type=2 (image)
  tdm_g1_t g1;
  g1[0] = (int)(1u << 16);                     // workgroup_mask=0, data_size=1 (2 bytes)
  g1[1] = (int)((tw & 0xFFFFu) << 16);         // tensor_dim0[15:0]
  g1[2] = (int)((tw >> 16) | ((th_ & 0xFFFFu) << 16));   // tensor_dim0 hi | tensor_dim1 lo
  g1[3] = (int)((th_ >> 16) | ((tw & 0xFFFFu) << 16));   // tensor_dim1 hi | tile_dim0
  g1[4] = (int)(th_ & 0xFFFFu);                // tile_dim1 | tile_dim2=0
  g1[5] = (int)stride_elems;                   // tensor_dim0_stride[31:0]
  g1[6] = 0;                                   // stride hi | tensor_dim1_stride lo
  g1[7] = 0;
  tdm_g2_t z4 = {0, 0, 0, 0};                  // groups 2/3 unused (<=2D tensor)
  tdm_g1_t z8 = {0, 0, 0, 0, 0, 0, 0, 0};
  __builtin_amdgcn_tensor_load_to_lds(g0, g1, z4, z4, z8, 0);
}
#endif

// ---------------- helpers ----------------

__device__ __forceinline__ bf16_t f2bf(float f) {
  unsigned u = __builtin_bit_cast(unsigned, f);
  u = (u + 0x7FFFu + ((u >> 16) & 1u)) >> 16;   // round-to-nearest-even
  unsigned short s = (unsigned short)u;
  return __builtin_bit_cast(bf16_t, s);
}

__device__ __forceinline__ v16bf cat16(v8bf lo, v8bf hi) {
  return __builtin_shufflevector(lo, hi, 0,1,2,3,4,5,6,7,8,9,10,11,12,13,14,15);
}

// A-matrix 16x32 bf16 fragment (contiguous k, matches ldb_frag):
//   lanes 0-15 : M=lane,    halves -> K=0..15
//   lanes16-31 : M=lane-16, halves -> K=16..31
__device__ __forceinline__ v16bf lda_frag(const bf16_t* row, int lane) {
  int off = (lane < 16) ? 0 : 16;
  v8bf lo = *(const v8bf*)(row + off);
  v8bf hi = *(const v8bf*)(row + off + 8);
  return cat16(lo, hi);
}

// B-matrix 32x16 bf16 fragment:
//   lanes 0-15 : N=lane,    halves -> K=0..15
//   lanes16-31 : N=lane-16, halves -> K=16..31
// `rowk` points at element [n][k_base] of an [N][K] row-major tile.
__device__ __forceinline__ v16bf ldb_frag(const bf16_t* rowk, int lane) {
  int off = (lane < 16) ? 0 : 16;
  v8bf lo = *(const v8bf*)(rowk + off);
  v8bf hi = *(const v8bf*)(rowk + off + 8);
  return cat16(lo, hi);
}

__device__ __forceinline__ v8f wmma_bf16(v16bf a, v16bf b, v8f c) {
  return __builtin_amdgcn_wmma_f32_16x16x32_bf16(false, a, false, b, (short)0, c,
                                                 false, false);
}

// T5 bidirectional relative position bucket (32 buckets -> half=16, max_exact=8,
// MAX_DISTANCE=128 -> log(16)=2.7725887...)
__device__ __forceinline__ int rel_bucket(int rp) {
  int rb = (rp > 0) ? 16 : 0;
  int a  = (rp < 0) ? -rp : rp;
  int idx;
  if (a < 8) {
    idx = a;
  } else {
    float l = __logf((float)a * 0.125f) * (8.0f / 2.7725887222397812f);
    idx = 8 + (int)l;
    idx = (idx < 15) ? idx : 15;
  }
  return rb + idx;
}

// ---------------- kernels ----------------

__global__ void cvt_kernel(const float* __restrict__ src, bf16_t* __restrict__ dst, int n) {
  int i = blockIdx.x * blockDim.x + threadIdx.x;
  if (i < n) dst[i] = f2bf(src[i]);
}

// Y = X * W^T, X[M][K] bf16, W[N][K] bf16 row-major.
// Block 256 threads = 8 waves; tile 128(M) x 64(N), K-step 64. LDS staged via TDM.
// Output bf16 head-major [B][H][S][D_KV].
__global__ __launch_bounds__(256) void gemm_qkv_kernel(
    const bf16_t* __restrict__ X,
    const bf16_t* __restrict__ Wq, const bf16_t* __restrict__ Wk,
    const bf16_t* __restrict__ Wv,
    bf16_t* __restrict__ Qb, bf16_t* __restrict__ Kb, bf16_t* __restrict__ Vb) {
  __shared__ __align__(16) bf16_t As[128 * 64];   // [m][k]
  __shared__ __align__(16) bf16_t Bs[64 * 64];    // [n][k]
  const int which = blockIdx.z;
  const bf16_t* W = (which == 0) ? Wq : (which == 1) ? Wk : Wv;
  bf16_t* Out     = (which == 0) ? Qb : (which == 1) ? Kb : Vb;
  const int n0 = blockIdx.x * 64;
  const int m0 = blockIdx.y * 128;
  const int t = threadIdx.x;
  const int lane = t & 31, wave = t >> 5;
  const int col = lane & 15, half8 = (lane < 16) ? 0 : 8;

  v8f acc[4] = {};
  for (int k0 = 0; k0 < D_MODEL; k0 += 64) {
    __syncthreads();   // prior-iteration LDS reads complete
#if HAVE_TDM
    if (wave == 0) {
      tdm_load_2d(As, X + (size_t)m0 * D_MODEL + k0, 64, 128, D_MODEL);
      tdm_load_2d(Bs, W + (size_t)n0 * D_MODEL + k0, 64, 64, D_MODEL);
      __builtin_amdgcn_s_wait_tensorcnt(0);
    }
#else
    {
      int m = t >> 1, ko = (t & 1) * 32;
      const bf16_t* src = X + (size_t)(m0 + m) * D_MODEL + k0 + ko;
      v8bf* dst = (v8bf*)&As[m * 64 + ko];
      dst[0] = *(const v8bf*)(src);
      dst[1] = *(const v8bf*)(src + 8);
      dst[2] = *(const v8bf*)(src + 16);
      dst[3] = *(const v8bf*)(src + 24);
    }
    {
      int n = t >> 2, ko = (t & 3) * 16;
      const bf16_t* src = W + (size_t)(n0 + n) * D_MODEL + k0 + ko;
      v8bf* dst = (v8bf*)&Bs[n * 64 + ko];
      dst[0] = *(const v8bf*)(src);
      dst[1] = *(const v8bf*)(src + 8);
    }
#endif
    __syncthreads();
    v16bf a0 = lda_frag(&As[(wave * 16 + col) * 64 + 0],  lane);
    v16bf a1 = lda_frag(&As[(wave * 16 + col) * 64 + 32], lane);
#pragma unroll
    for (int nt = 0; nt < 4; ++nt) {
      acc[nt] = wmma_bf16(a0, ldb_frag(&Bs[(nt * 16 + col) * 64 + 0],  lane), acc[nt]);
      acc[nt] = wmma_bf16(a1, ldb_frag(&Bs[(nt * 16 + col) * 64 + 32], lane), acc[nt]);
    }
  }
#pragma unroll
  for (int nt = 0; nt < 4; ++nt) {
    int ng = n0 + nt * 16 + col;
    int h = ng >> 6, d = ng & 63;
#pragma unroll
    for (int v = 0; v < 8; ++v) {
      int mg = m0 + wave * 16 + half8 + v;
      int bb = mg >> 11, s = mg & 2047;
      Out[(((size_t)(bb * N_HEADS + h)) * SEQ + s) * D_KV + d] = f2bf(acc[nt][v]);
    }
  }
}

// Flash-attention: one block per (128 q-rows, head, batch). 8 waves x 16 q-rows.
__global__ __launch_bounds__(256) void attn_kernel(
    const bf16_t* __restrict__ Qb, const bf16_t* __restrict__ Kb,
    const bf16_t* __restrict__ Vb, const int* __restrict__ positions,
    const float* __restrict__ mask, const float* __restrict__ rel_bias,
    bf16_t* __restrict__ Ctx) {
  __shared__ __align__(16) bf16_t Qs[128 * 64];
  __shared__ __align__(16) bf16_t Ks[64 * 64];
  __shared__ __align__(16) bf16_t Vs[64 * 64];
  __shared__ __align__(16) bf16_t Ps[8][16 * 64];   // per-wave P staging (D->A relayout)

  const int q0 = blockIdx.x * 128;
  const int h  = blockIdx.y;
  const int b  = blockIdx.z;
  const int t = threadIdx.x, lane = t & 31, wave = t >> 5;
  const int col = lane & 15, half8 = (lane < 16) ? 0 : 8;

  const size_t head = (size_t)(b * N_HEADS + h) * SEQ * D_KV;
  const bf16_t* Qh = Qb + head;
  const bf16_t* Kh = Kb + head;
  const bf16_t* Vh = Vb + head;

#if HAVE_TDM
  if (wave == 0) tdm_load_2d(Qs, Qh + (size_t)q0 * D_KV, 128 * 64, 1, 128 * 64);
#else
  {
    const v8bf* src = (const v8bf*)(Qh + (size_t)q0 * D_KV);
    v8bf* dst = (v8bf*)Qs;
    for (int i = t; i < (128 * 64) / 8; i += 256) dst[i] = src[i];
  }
#endif

  float mrow[8], lrow[8], posq[8];
  v8f O[4] = {};
#pragma unroll
  for (int v = 0; v < 8; ++v) {
    mrow[v] = -3.0e38f;
    lrow[v] = 0.f;
    posq[v] = (float)positions[b * SEQ + q0 + wave * 16 + half8 + v];
  }

  for (int kt = 0; kt < SEQ; kt += 64) {
    __syncthreads();   // prior-iteration LDS reads complete
#if HAVE_TDM
    if (wave == 0) {
      tdm_load_2d(Ks, Kh + (size_t)kt * D_KV, 64 * 64, 1, 64 * 64);
      tdm_load_2d(Vs, Vh + (size_t)kt * D_KV, 64 * 64, 1, 64 * 64);
      __builtin_amdgcn_s_wait_tensorcnt(0);   // also covers the Q-tile DMA
    }
#else
    {
      const v8bf* ksrc = (const v8bf*)(Kh + (size_t)kt * D_KV);
      const v8bf* vsrc = (const v8bf*)(Vh + (size_t)kt * D_KV);
      v8bf* kdst = (v8bf*)Ks;
      v8bf* vdst = (v8bf*)Vs;
      for (int i = t; i < (64 * 64) / 8; i += 256) {
        kdst[i] = ksrc[i];
        vdst[i] = vsrc[i];
      }
    }
#endif
    __syncthreads();

    // ---- scores S = Q * K^T  (d_kv = 64 -> two WMMA per 16x16 score tile)
    v16bf a0 = lda_frag(&Qs[(wave * 16 + col) * D_KV + 0],  lane);
    v16bf a1 = lda_frag(&Qs[(wave * 16 + col) * D_KV + 32], lane);
    v8f S[4] = {};
#pragma unroll
    for (int nt = 0; nt < 4; ++nt) {
      S[nt] = wmma_bf16(a0, ldb_frag(&Ks[(nt * 16 + col) * D_KV + 0],  lane), S[nt]);
      S[nt] = wmma_bf16(a1, ldb_frag(&Ks[(nt * 16 + col) * D_KV + 32], lane), S[nt]);
    }

    // ---- position bias + mask term
#pragma unroll
    for (int nt = 0; nt < 4; ++nt) {
      int kg = kt + nt * 16 + col;
      float pk = (float)positions[b * SEQ + kg];
      float mterm = -1000.0f * (1.0f - mask[b * SEQ + kg]);
#pragma unroll
      for (int v = 0; v < 8; ++v) {
        int rp = (int)(pk - posq[v]);
        S[nt][v] += rel_bias[rel_bucket(rp) * N_HEADS + h] + mterm;
      }
    }

    // ---- online softmax (one score row spans a 16-lane half of the wave)
#pragma unroll
    for (int v = 0; v < 8; ++v) {
      float rmax = fmaxf(fmaxf(S[0][v], S[1][v]), fmaxf(S[2][v], S[3][v]));
#pragma unroll
      for (int off = 8; off >= 1; off >>= 1)
        rmax = fmaxf(rmax, __shfl_xor(rmax, off, 16));
      float mnew  = fmaxf(mrow[v], rmax);
      float scale = __expf(mrow[v] - mnew);
      mrow[v] = mnew;
      float rsum = 0.f;
#pragma unroll
      for (int nt = 0; nt < 4; ++nt) {
        float p = __expf(S[nt][v] - mnew);
        S[nt][v] = p;
        rsum += p;
      }
#pragma unroll
      for (int off = 8; off >= 1; off >>= 1) rsum += __shfl_xor(rsum, off, 16);
      lrow[v] = lrow[v] * scale + rsum;
#pragma unroll
      for (int nt = 0; nt < 4; ++nt) O[nt][v] *= scale;
    }

    // ---- stage P (C/D layout -> A layout) through per-wave LDS
    bf16_t* Pw = &Ps[wave][0];
#pragma unroll
    for (int nt = 0; nt < 4; ++nt)
#pragma unroll
      for (int v = 0; v < 8; ++v)
        Pw[(half8 + v) * 64 + nt * 16 + col] = f2bf(S[nt][v]);

    v16bf p0 = lda_frag(&Pw[col * 64 + 0],  lane);
    v16bf p1 = lda_frag(&Pw[col * 64 + 32], lane);

    // ---- O += P * V   (B operand gathered: B[k][n] = Vs[k][d])
    int koff = (lane < 16) ? 0 : 16;
#pragma unroll
    for (int nt = 0; nt < 4; ++nt) {
      v16bf vb0, vb1;
#pragma unroll
      for (int hh = 0; hh < 16; ++hh) {
        vb0[hh] = Vs[(koff + hh) * D_KV + nt * 16 + col];
        vb1[hh] = Vs[(32 + koff + hh) * D_KV + nt * 16 + col];
      }
      O[nt] = wmma_bf16(p0, vb0, O[nt]);
      O[nt] = wmma_bf16(p1, vb1, O[nt]);
    }
  }

  // ---- normalize, write context [B][S][H*D_KV] bf16
#pragma unroll
  for (int v = 0; v < 8; ++v) {
    float inv = 1.0f / lrow[v];
    int sq = q0 + wave * 16 + half8 + v;
    size_t base = ((size_t)(b * SEQ + sq)) * INNER + (size_t)h * D_KV;
#pragma unroll
    for (int nt = 0; nt < 4; ++nt)
      Ctx[base + nt * 16 + col] = f2bf(O[nt][v] * inv);
  }
}

// Out = Ctx * Wo^T, f32 output row-major [ROWS][D_MODEL]
__global__ __launch_bounds__(256) void gemm_out_kernel(
    const bf16_t* __restrict__ A, const bf16_t* __restrict__ W,
    float* __restrict__ Out) {
  __shared__ __align__(16) bf16_t As[128 * 64];
  __shared__ __align__(16) bf16_t Bs[64 * 64];
  const int n0 = blockIdx.x * 64;
  const int m0 = blockIdx.y * 128;
  const int t = threadIdx.x;
  const int lane = t & 31, wave = t >> 5;
  const int col = lane & 15, half8 = (lane < 16) ? 0 : 8;

  v8f acc[4] = {};
  for (int k0 = 0; k0 < INNER; k0 += 64) {
    __syncthreads();
#if HAVE_TDM
    if (wave == 0) {
      tdm_load_2d(As, A + (size_t)m0 * INNER + k0, 64, 128, INNER);
      tdm_load_2d(Bs, W + (size_t)n0 * INNER + k0, 64, 64, INNER);
      __builtin_amdgcn_s_wait_tensorcnt(0);
    }
#else
    {
      int m = t >> 1, ko = (t & 1) * 32;
      const bf16_t* src = A + (size_t)(m0 + m) * INNER + k0 + ko;
      v8bf* dst = (v8bf*)&As[m * 64 + ko];
      dst[0] = *(const v8bf*)(src);
      dst[1] = *(const v8bf*)(src + 8);
      dst[2] = *(const v8bf*)(src + 16);
      dst[3] = *(const v8bf*)(src + 24);
    }
    {
      int n = t >> 2, ko = (t & 3) * 16;
      const bf16_t* src = W + (size_t)(n0 + n) * INNER + k0 + ko;
      v8bf* dst = (v8bf*)&Bs[n * 64 + ko];
      dst[0] = *(const v8bf*)(src);
      dst[1] = *(const v8bf*)(src + 8);
    }
#endif
    __syncthreads();
    v16bf a0 = lda_frag(&As[(wave * 16 + col) * 64 + 0],  lane);
    v16bf a1 = lda_frag(&As[(wave * 16 + col) * 64 + 32], lane);
#pragma unroll
    for (int nt = 0; nt < 4; ++nt) {
      acc[nt] = wmma_bf16(a0, ldb_frag(&Bs[(nt * 16 + col) * 64 + 0],  lane), acc[nt]);
      acc[nt] = wmma_bf16(a1, ldb_frag(&Bs[(nt * 16 + col) * 64 + 32], lane), acc[nt]);
    }
  }
#pragma unroll
  for (int nt = 0; nt < 4; ++nt) {
    int ng = n0 + nt * 16 + col;
#pragma unroll
    for (int v = 0; v < 8; ++v) {
      int mg = m0 + wave * 16 + half8 + v;
      Out[(size_t)mg * D_MODEL + ng] = acc[nt][v];
    }
  }
}

// ---------------- launch ----------------

extern "C" void kernel_launch(void* const* d_in, const int* in_sizes, int n_in,
                              void* d_out, int out_size, void* d_ws, size_t ws_size,
                              hipStream_t stream) {
  (void)in_sizes; (void)n_in; (void)out_size; (void)ws_size;
  const float* hidden    = (const float*)d_in[0];
  const int*   positions = (const int*)d_in[1];
  const float* mask      = (const float*)d_in[2];
  const float* wq        = (const float*)d_in[3];
  const float* wk        = (const float*)d_in[4];
  const float* wv        = (const float*)d_in[5];
  const float* wo        = (const float*)d_in[6];
  const float* rel_bias  = (const float*)d_in[7];
  float* out = (float*)d_out;

  char* p = (char*)d_ws;
  auto alloc = [&](size_t elems) -> bf16_t* {
    bf16_t* r = (bf16_t*)p;
    p += elems * sizeof(bf16_t);
    return r;
  };
  bf16_t* Xb  = alloc((size_t)ROWS * D_MODEL);
  bf16_t* Wqb = alloc((size_t)INNER * D_MODEL);
  bf16_t* Wkb = alloc((size_t)INNER * D_MODEL);
  bf16_t* Wvb = alloc((size_t)INNER * D_MODEL);
  bf16_t* Wob = alloc((size_t)D_MODEL * INNER);
  bf16_t* Qb  = alloc((size_t)ROWS * INNER);
  bf16_t* Kb  = alloc((size_t)ROWS * INNER);
  bf16_t* Vb  = alloc((size_t)ROWS * INNER);
  bf16_t* Ctx = alloc((size_t)ROWS * INNER);

  const int nX = ROWS * D_MODEL;
  const int nW = INNER * D_MODEL;
  cvt_kernel<<<(nX + 255) / 256, 256, 0, stream>>>(hidden, Xb, nX);
  cvt_kernel<<<(nW + 255) / 256, 256, 0, stream>>>(wq, Wqb, nW);
  cvt_kernel<<<(nW + 255) / 256, 256, 0, stream>>>(wk, Wkb, nW);
  cvt_kernel<<<(nW + 255) / 256, 256, 0, stream>>>(wv, Wvb, nW);
  cvt_kernel<<<(nW + 255) / 256, 256, 0, stream>>>(wo, Wob, nW);

  gemm_qkv_kernel<<<dim3(INNER / 64, ROWS / 128, 3), 256, 0, stream>>>(
      Xb, Wqb, Wkb, Wvb, Qb, Kb, Vb);

  attn_kernel<<<dim3(SEQ / 128, N_HEADS, BATCH), 256, 0, stream>>>(
      Qb, Kb, Vb, positions, mask, rel_bias, Ctx);

  gemm_out_kernel<<<dim3(D_MODEL / 64, ROWS / 128), 256, 0, stream>>>(Ctx, Wob, out);
}
